// MultiHeadAttention_22660247454093
// MI455X (gfx1250) — matrix operand order repaired and checked by reading, not compiled, not run on hardware
//
#include <hip/hip_runtime.h>
#include <stdint.h>

#define D_MODEL 1024
#define NUM_HEADS 16
#define DK 64
#define SEQ 2048
#define BATCH 2

typedef __bf16 bf16_t;
typedef __attribute__((ext_vector_type(16))) __bf16 v16bf;
typedef __attribute__((ext_vector_type(8)))  __bf16 v8bf;
typedef __attribute__((ext_vector_type(4)))  __bf16 v4bf;
typedef __attribute__((ext_vector_type(8)))  float  v8f;
typedef __attribute__((ext_vector_type(4)))  int    v4i;

typedef __attribute__((address_space(1))) v4i as1_v4i;   // global int4
typedef __attribute__((address_space(3))) v4i as3_v4i;   // LDS int4

#if __has_builtin(__builtin_amdgcn_global_load_async_to_lds_b128)
#define HAVE_ASYNC_LDS 1
#else
#define HAVE_ASYNC_LDS 0
#endif

static __device__ __forceinline__ void wait_async_zero() {
#if __has_builtin(__builtin_amdgcn_s_wait_asynccnt)
  __builtin_amdgcn_s_wait_asynccnt(0);
#else
  asm volatile("s_wait_asynccnt 0x0" ::: "memory");
#endif
}

union ABfrag { v16bf v; v8bf h[2]; };

static __device__ __forceinline__ bf16_t f2bf(float f) {
  union { float f; uint32_t u; } in; in.f = f;
  uint32_t u = in.u;
  u += 0x7fffu + ((u >> 16) & 1u);   // round-to-nearest-even
  union { uint16_t s; bf16_t b; } out; out.s = (uint16_t)(u >> 16);
  return out.b;
}

// ---------------------------------------------------------------------------
// fp32 -> bf16 bulk conversion (vectorized: 16B in, 8B out per thread)
// ---------------------------------------------------------------------------
__global__ __launch_bounds__(256)
void cvt_f32_bf16(const float* __restrict__ src, bf16_t* __restrict__ dst, int n) {
  int i = (blockIdx.x * 256 + threadIdx.x) * 4;
  if (i + 3 < n) {
    float4 f = *reinterpret_cast<const float4*>(src + i);
    v4bf o;
    o[0] = f2bf(f.x); o[1] = f2bf(f.y); o[2] = f2bf(f.z); o[3] = f2bf(f.w);
    *reinterpret_cast<v4bf*>(dst + i) = o;
  }
}

// ---------------------------------------------------------------------------
// GEMM: out = A(MxK, bf16, row-major) * W(NxK, bf16, row-major)^T + bias
// MODE 0: store bf16 row-major [M x N]
// MODE 1: store bf16 transposed per head: [b, h, d, s]  (for V)
// MODE 2: store fp32 row-major [M x N]   (final projection)
// Block tile 128x128, 8 waves of 32(M) x 64(N), BK = 32.
// Tile staging uses GLOBAL_LOAD_ASYNC_TO_LDS_B128 (ASYNCcnt) when available.
// ---------------------------------------------------------------------------
template<int MODE>
__global__ __launch_bounds__(256)
void gemm_bf16_wmma(const bf16_t* __restrict__ A, const bf16_t* __restrict__ W,
                    const float* __restrict__ bias, void* __restrict__ outp,
                    int M, int N, int K) {
  constexpr int BM = 128, BN = 128, BK = 32, LDT = BK + 8;   // padded LDS stride
  __shared__ bf16_t As[BM * LDT];
  __shared__ bf16_t Bs[BN * LDT];

  const int t    = threadIdx.x;
  const int lane = t & 31;
  const int wid  = t >> 5;
  const int half = lane >> 4;
  const int l16  = lane & 15;
  const int wm   = wid & 3;       // 0..3 -> M offset 32*wm
  const int wn   = wid >> 2;      // 0..1 -> N offset 64*wn
  const int mBase = blockIdx.y * BM;
  const int nBase = blockIdx.x * BN;

  const v8f vzero = {0.f,0.f,0.f,0.f,0.f,0.f,0.f,0.f};
  v8f acc[2][4];
#pragma unroll
  for (int i = 0; i < 2; ++i)
#pragma unroll
    for (int j = 0; j < 4; ++j) acc[i][j] = vzero;

  const int loadRow = t >> 1;           // 0..127
  const int loadCol = (t & 1) * 16;     // 0 or 16

  for (int k0 = 0; k0 < K; k0 += BK) {
    __syncthreads();
    {
      const bf16_t* sa = A + (size_t)(mBase + loadRow) * K + k0 + loadCol;
      const bf16_t* sb = W + (size_t)(nBase + loadRow) * K + k0 + loadCol;
#if HAVE_ASYNC_LDS
      __builtin_amdgcn_global_load_async_to_lds_b128(
          (as1_v4i*)sa,       (as3_v4i*)&As[loadRow * LDT + loadCol],     0, 0);
      __builtin_amdgcn_global_load_async_to_lds_b128(
          (as1_v4i*)(sa + 8), (as3_v4i*)&As[loadRow * LDT + loadCol + 8], 0, 0);
      __builtin_amdgcn_global_load_async_to_lds_b128(
          (as1_v4i*)sb,       (as3_v4i*)&Bs[loadRow * LDT + loadCol],     0, 0);
      __builtin_amdgcn_global_load_async_to_lds_b128(
          (as1_v4i*)(sb + 8), (as3_v4i*)&Bs[loadRow * LDT + loadCol + 8], 0, 0);
      wait_async_zero();
#else
      *reinterpret_cast<v8bf*>(&As[loadRow * LDT + loadCol])     = *reinterpret_cast<const v8bf*>(sa);
      *reinterpret_cast<v8bf*>(&As[loadRow * LDT + loadCol + 8]) = *reinterpret_cast<const v8bf*>(sa + 8);
      *reinterpret_cast<v8bf*>(&Bs[loadRow * LDT + loadCol])     = *reinterpret_cast<const v8bf*>(sb);
      *reinterpret_cast<v8bf*>(&Bs[loadRow * LDT + loadCol + 8]) = *reinterpret_cast<const v8bf*>(sb + 8);
#endif
      if (k0 + BK < K) {  // prefetch next K tile -> global_prefetch_b8
        __builtin_prefetch(A + (size_t)(mBase + loadRow) * K + k0 + BK + loadCol, 0, 1);
        __builtin_prefetch(W + (size_t)(nBase + loadRow) * K + k0 + BK + loadCol, 0, 1);
      }
    }
    __syncthreads();

    ABfrag af[2], bf[4];
#pragma unroll
    for (int fm = 0; fm < 2; ++fm) {
      int m = wm * 32 + fm * 16 + l16;
      af[fm].h[0] = *reinterpret_cast<const v8bf*>(&As[m * LDT + 8 * half]);
      af[fm].h[1] = *reinterpret_cast<const v8bf*>(&As[m * LDT + 16 + 8 * half]);
    }
#pragma unroll
    for (int fn = 0; fn < 4; ++fn) {
      int n = wn * 64 + fn * 16 + l16;
      bf[fn].h[0] = *reinterpret_cast<const v8bf*>(&Bs[n * LDT + 8 * half]);
      bf[fn].h[1] = *reinterpret_cast<const v8bf*>(&Bs[n * LDT + 16 + 8 * half]);
    }
#pragma unroll
    for (int fm = 0; fm < 2; ++fm)
#pragma unroll
      for (int fn = 0; fn < 4; ++fn)
        acc[fm][fn] = __builtin_amdgcn_wmma_f32_16x16x32_bf16(
            false, af[fm].v, false, bf[fn].v, (short)0, acc[fm][fn], false, false);
  }

  // Epilogue: C/D layout -> lane holds col N = l16, rows M = r + 8*half
#pragma unroll
  for (int fn = 0; fn < 4; ++fn) {
    const int nG = nBase + wn * 64 + fn * 16 + l16;
    const float bv = bias[nG];
#pragma unroll
    for (int fm = 0; fm < 2; ++fm) {
      const int mG0 = mBase + wm * 32 + fm * 16 + 8 * half;
#pragma unroll
      for (int r = 0; r < 8; ++r) {
        const float val = acc[fm][fn][r] + bv;
        const int m = mG0 + r;
        if constexpr (MODE == 2) {
          reinterpret_cast<float*>(outp)[(size_t)m * N + nG] = val;
        } else if constexpr (MODE == 0) {
          reinterpret_cast<bf16_t*>(outp)[(size_t)m * N + nG] = f2bf(val);
        } else {
          const int b = m / SEQ, s = m % SEQ;
          const int h = nG / DK, d = nG % DK;
          reinterpret_cast<bf16_t*>(outp)[(((size_t)(b * NUM_HEADS + h)) * DK + d) * SEQ + s] = f2bf(val);
        }
      }
    }
  }
}

// ---------------------------------------------------------------------------
// Flash attention: per block one (b, h) and 128 q rows (8 waves x 16 rows).
// Q layout [b, s, h*64+d] bf16; K layout same; V transposed [b, h, d, s] bf16.
// Output O: [b, s, h*64+d] bf16.
// ---------------------------------------------------------------------------
__global__ __launch_bounds__(256)
void attn_flash_wmma(const bf16_t* __restrict__ Q, const bf16_t* __restrict__ Kp,
                     const bf16_t* __restrict__ Vt, bf16_t* __restrict__ O) {
  constexpr int LKT = DK + 8;   // 72: K tile row stride (row = key j, 64 d)
  constexpr int LVT = 32 + 8;   // 40: Vt tile row stride (row = d, 32 j)
  constexpr int LPT = 32 + 8;   // 40: P tile row stride (row = q, 32 j)
  __shared__ bf16_t Ks[32 * LKT];
  __shared__ bf16_t Vs[DK * LVT];
  __shared__ bf16_t Ps[8 * 16 * LPT];

  const int t    = threadIdx.x;
  const int lane = t & 31;
  const int wid  = t >> 5;
  const int half = lane >> 4;
  const int l16  = lane & 15;
  const int b    = blockIdx.z;
  const int h    = blockIdx.y;
  const int q0   = blockIdx.x * 128 + wid * 16;
  bf16_t* Pw = &Ps[wid * 16 * LPT];

  // Q fragments (16 q rows x 64 d) held in registers for the whole pass
  ABfrag qf[2];
  {
    const bf16_t* qp = Q + ((size_t)(b * SEQ) + q0 + l16) * D_MODEL + h * DK;
#pragma unroll
    for (int kc = 0; kc < 2; ++kc) {
      qf[kc].h[0] = *reinterpret_cast<const v8bf*>(qp + kc * 32 + 8 * half);
      qf[kc].h[1] = *reinterpret_cast<const v8bf*>(qp + kc * 32 + 16 + 8 * half);
    }
  }

  const v8f vzero = {0.f,0.f,0.f,0.f,0.f,0.f,0.f,0.f};
  v8f o[4];
#pragma unroll
  for (int fn = 0; fn < 4; ++fn) o[fn] = vzero;
  float mrow[8], lrow[8];
#pragma unroll
  for (int r = 0; r < 8; ++r) { mrow[r] = -3.402823466e38f; lrow[r] = 0.f; }

  const float scale = 0.125f;   // 1/sqrt(64)

  for (int j0 = 0; j0 < SEQ; j0 += 32) {
    __syncthreads();
    {   // cooperative stage of K tile (32 x 64) and Vt tile (64 x 32)
      const int jr = t >> 3, dc = (t & 7) * 8;
      const bf16_t* ksrc = Kp + ((size_t)(b * SEQ) + j0 + jr) * D_MODEL + h * DK + dc;
      const int dr = t >> 2, jc = (t & 3) * 8;
      const bf16_t* vsrc = Vt + (((size_t)(b * NUM_HEADS + h)) * DK + dr) * SEQ + j0 + jc;
#if HAVE_ASYNC_LDS
      __builtin_amdgcn_global_load_async_to_lds_b128(
          (as1_v4i*)ksrc, (as3_v4i*)&Ks[jr * LKT + dc], 0, 0);
      __builtin_amdgcn_global_load_async_to_lds_b128(
          (as1_v4i*)vsrc, (as3_v4i*)&Vs[dr * LVT + jc], 0, 0);
      wait_async_zero();
#else
      *reinterpret_cast<v8bf*>(&Ks[jr * LKT + dc]) = *reinterpret_cast<const v8bf*>(ksrc);
      *reinterpret_cast<v8bf*>(&Vs[dr * LVT + jc]) = *reinterpret_cast<const v8bf*>(vsrc);
#endif
    }
    __syncthreads();

    // ---- scores S = Q K^T : two 16x16 j-halves, each accumulated over d=64
    v8f s[2];
#pragma unroll
    for (int jh = 0; jh < 2; ++jh) {
      const int n = jh * 16 + l16;          // key index within tile
      ABfrag kf0, kf1;
      kf0.h[0] = *reinterpret_cast<const v8bf*>(&Ks[n * LKT + 8 * half]);
      kf0.h[1] = *reinterpret_cast<const v8bf*>(&Ks[n * LKT + 16 + 8 * half]);
      kf1.h[0] = *reinterpret_cast<const v8bf*>(&Ks[n * LKT + 32 + 8 * half]);
      kf1.h[1] = *reinterpret_cast<const v8bf*>(&Ks[n * LKT + 32 + 16 + 8 * half]);
      v8f z = vzero;
      z = __builtin_amdgcn_wmma_f32_16x16x32_bf16(false, qf[0].v, false, kf0.v, (short)0, z, false, false);
      z = __builtin_amdgcn_wmma_f32_16x16x32_bf16(false, qf[1].v, false, kf1.v, (short)0, z, false, false);
      s[jh] = z;
    }

    // ---- online softmax (rows r+8*half live across the 16-lane half-groups)
#pragma unroll
    for (int r = 0; r < 8; ++r) {
      const float v0 = s[0][r] * scale;
      const float v1 = s[1][r] * scale;
      float mx = fmaxf(v0, v1);
      mx = fmaxf(mx, __shfl_xor(mx, 1, 16));
      mx = fmaxf(mx, __shfl_xor(mx, 2, 16));
      mx = fmaxf(mx, __shfl_xor(mx, 4, 16));
      mx = fmaxf(mx, __shfl_xor(mx, 8, 16));
      const float mnew  = fmaxf(mrow[r], mx);
      const float alpha = __expf(mrow[r] - mnew);
      const float p0 = __expf(v0 - mnew);
      const float p1 = __expf(v1 - mnew);
      float sum = p0 + p1;
      sum += __shfl_xor(sum, 1, 16);
      sum += __shfl_xor(sum, 2, 16);
      sum += __shfl_xor(sum, 4, 16);
      sum += __shfl_xor(sum, 8, 16);
      mrow[r] = mnew;
      lrow[r] = lrow[r] * alpha + sum;
      // transpose-stage P into per-wave LDS scratch ([q_local][j_local])
      Pw[(r + 8 * half) * LPT + l16]      = f2bf(p0);
      Pw[(r + 8 * half) * LPT + 16 + l16] = f2bf(p1);
#pragma unroll
      for (int fn = 0; fn < 4; ++fn) o[fn][r] *= alpha;
    }

    // ---- O += P @ V  (A-frag of P from LDS, B-frags from transposed V tile)
    ABfrag pf;
    pf.h[0] = *reinterpret_cast<const v8bf*>(&Pw[l16 * LPT + 8 * half]);
    pf.h[1] = *reinterpret_cast<const v8bf*>(&Pw[l16 * LPT + 16 + 8 * half]);
#pragma unroll
    for (int fn = 0; fn < 4; ++fn) {
      const int d = fn * 16 + l16;
      ABfrag vf;
      vf.h[0] = *reinterpret_cast<const v8bf*>(&Vs[d * LVT + 8 * half]);
      vf.h[1] = *reinterpret_cast<const v8bf*>(&Vs[d * LVT + 16 + 8 * half]);
      o[fn] = __builtin_amdgcn_wmma_f32_16x16x32_bf16(false, pf.v, false, vf.v, (short)0, o[fn], false, false);
    }
  }

  // ---- epilogue: normalize and store bf16 [b, s, h*64+d]
#pragma unroll
  for (int r = 0; r < 8; ++r) {
    const float inv = 1.0f / lrow[r];
    const int srow = q0 + r + 8 * half;
#pragma unroll
    for (int fn = 0; fn < 4; ++fn) {
      O[((size_t)(b * SEQ) + srow) * D_MODEL + h * DK + fn * 16 + l16] =
          f2bf(o[fn][r] * inv);
    }
  }
}

// ---------------------------------------------------------------------------
extern "C" void kernel_launch(void* const* d_in, const int* in_sizes, int n_in,
                              void* d_out, int out_size, void* d_ws, size_t ws_size,
                              hipStream_t stream) {
  const float* query = (const float*)d_in[0];
  const float* key   = (const float*)d_in[1];
  const float* value = (const float*)d_in[2];
  const float* Wq    = (const float*)d_in[3];
  const float* bq    = (const float*)d_in[4];
  const float* Wk    = (const float*)d_in[5];
  const float* bk    = (const float*)d_in[6];
  const float* Wv    = (const float*)d_in[7];
  const float* bv    = (const float*)d_in[8];
  const float* Wo    = (const float*)d_in[9];
  const float* bo    = (const float*)d_in[10];

  const size_t NX = (size_t)BATCH * SEQ * D_MODEL;   // 4 Mi elems
  const size_t NW = (size_t)D_MODEL * D_MODEL;       // 1 Mi elems

  bf16_t* w   = (bf16_t*)d_ws;
  bf16_t* Xq  = w; w += NX;
  bf16_t* Xk  = w; w += NX;
  bf16_t* Xv  = w; w += NX;
  bf16_t* Wqb = w; w += NW;
  bf16_t* Wkb = w; w += NW;
  bf16_t* Wvb = w; w += NW;
  bf16_t* Wob = w; w += NW;
  bf16_t* Qb  = w; w += NX;
  bf16_t* Kb  = w; w += NX;
  bf16_t* Vtb = w; w += NX;
  bf16_t* Ob  = w; w += NX;

  // 1) fp32 -> bf16 conversions
  cvt_f32_bf16<<<dim3((unsigned)(NX / 1024)), 256, 0, stream>>>(query, Xq, (int)NX);
  cvt_f32_bf16<<<dim3((unsigned)(NX / 1024)), 256, 0, stream>>>(key,   Xk, (int)NX);
  cvt_f32_bf16<<<dim3((unsigned)(NX / 1024)), 256, 0, stream>>>(value, Xv, (int)NX);
  cvt_f32_bf16<<<dim3((unsigned)(NW / 1024)), 256, 0, stream>>>(Wq, Wqb, (int)NW);
  cvt_f32_bf16<<<dim3((unsigned)(NW / 1024)), 256, 0, stream>>>(Wk, Wkb, (int)NW);
  cvt_f32_bf16<<<dim3((unsigned)(NW / 1024)), 256, 0, stream>>>(Wv, Wvb, (int)NW);
  cvt_f32_bf16<<<dim3((unsigned)(NW / 1024)), 256, 0, stream>>>(Wo, Wob, (int)NW);

  // 2) projections: Q, K row-major; V stored transposed per head
  const int M = BATCH * SEQ, N = D_MODEL, K = D_MODEL;
  dim3 gg(N / 128, M / 128);
  gemm_bf16_wmma<0><<<gg, 256, 0, stream>>>(Xq, Wqb, bq, Qb,  M, N, K);
  gemm_bf16_wmma<0><<<gg, 256, 0, stream>>>(Xk, Wkb, bk, Kb,  M, N, K);
  gemm_bf16_wmma<1><<<gg, 256, 0, stream>>>(Xv, Wvb, bv, Vtb, M, N, K);

  // 3) fused flash attention (per (b, h), 128 q rows per block)
  attn_flash_wmma<<<dim3(SEQ / 128, NUM_HEADS, BATCH), 256, 0, stream>>>(Qb, Kb, Vtb, Ob);

  // 4) output projection -> fp32 d_out
  gemm_bf16_wmma<2><<<gg, 256, 0, stream>>>(Ob, Wob, bo, d_out, M, N, K);
}
